// SurnameClassifier_85847806312669
// MI455X (gfx1250) — matrix operand order repaired and checked by reading, not compiled, not run on hardware
//
#include <hip/hip_runtime.h>
#include <math.h>

// SurnameClassifier for MI455X (gfx1250, wave32, WMMA).
//
// B=1024, S=512, E=128, H=256, V=128, C=18.
// Strategy:
//   1) xw_kernel:      xw[v][h] = emb[v]·W_ih[h] + b_ih[h] + b_hh[h]   (V x H f32)
//      -> turns the per-step input GEMM into a token-indexed table gather.
//   2) pack_whh:       W_hh (f32, [n][k]) -> f16 WMMA B-fragments (32x16 tiles),
//      lane-contiguous so each lane loads its 16 halves with one 32B load.
//   3) rnn_scan:       64 WGs x 256 thr. WG = 16 batch rows. Wave w owns N cols
//      [32w,32w+32). B-frags (2x8 v16h = 128 VGPRs) resident in registers for all
//      512 steps. h ping-pongs in LDS as f16. Per step: tokens loaded up-front
//      (overlap with GEMM), A-frags software-pipelined 2-deep over the 8 K-tiles,
//      16x v_wmma_f32_16x16x32_f16, batched xw gather, tanh, f16 writeback,
//      conditional last-hidden capture. One barrier per step.
//   4) head_kernel:    fp32 relu(y@fc1^T+b) @ fc2^T + b  (tiny).

typedef __attribute__((ext_vector_type(16))) _Float16 v16h;
typedef __attribute__((ext_vector_type(8)))  float    v8f;

#define B_  1024
#define S_  512
#define E_  128
#define H_  256
#define V_  128
#define C_  18

// ---------------------------------------------------------------------------
// Kernel 1: xw[v][h] = dot(emb[v,:], W_ih[h,:]) + b_ih[h] + b_hh[h]
// ---------------------------------------------------------------------------
__global__ void xw_kernel(const float* __restrict__ emb,
                          const float* __restrict__ W_ih,
                          const float* __restrict__ b_ih,
                          const float* __restrict__ b_hh,
                          float* __restrict__ xw) {
    __shared__ float se[E_];
    const int v = blockIdx.x;      // 0..127
    const int h = threadIdx.x;     // 0..255
    if (h < E_) se[h] = emb[v * E_ + h];
    __syncthreads();
    float acc = b_ih[h] + b_hh[h];
    const float* wrow = W_ih + h * E_;
#pragma unroll 4
    for (int k = 0; k < E_; ++k) acc += se[k] * wrow[k];
    xw[v * H_ + h] = acc;
}

// ---------------------------------------------------------------------------
// Kernel 2: pack W_hh[n][k] (f32) into f16 WMMA B-fragments.
// Fragment (nt,kt) covers n in [16nt,16nt+16), k in [32kt,32kt+32).
// Within a 32x16 (KxN) f16 B fragment (wave32, 8 VGPRs x 2 halves):
//   lane = n%16 + 16*((kin/8)%2);  vgpr = (kin%8)/2 + 4*(kin/16);  half = kin&1
// Stored lane-contiguous: frag*512 + lane*16 + vgpr*2 + half.
// ---------------------------------------------------------------------------
__global__ void pack_whh_kernel(const float* __restrict__ W_hh,
                                _Float16* __restrict__ packed) {
    const int idx = blockIdx.x * blockDim.x + threadIdx.x;   // 0..65535
    const int n   = idx >> 8;
    const int k   = idx & (H_ - 1);
    const int nt  = n >> 4,  nn  = n & 15;
    const int kt  = k >> 5,  kin = k & 31;
    const int lane = nn + ((kin & 8) ? 16 : 0);
    const int vg   = ((kin & 7) >> 1) + ((kin >> 4) << 2);
    const int hf   = kin & 1;
    const int dst  = ((nt * 8 + kt) * 512) + lane * 16 + vg * 2 + hf;
    packed[dst] = (_Float16)W_hh[n * H_ + k];   // B(k,n) = W_hh[n][k]
}

// ---------------------------------------------------------------------------
// Kernel 3: recurrent scan. Grid = 64, block = 256 (8 waves).
// ---------------------------------------------------------------------------
#define SM_XW_BYTES   (V_ * H_ * 4)          // 131072
#define SM_TOK_BYTES  (16 * S_ * 4)          //  32768
#define SM_LEN_BYTES  (16 * 4)               //     64
#define SM_H_BYTES    (2 * 16 * H_ * 2)      //  16384 (ping-pong f16)
#define SM_TOTAL      (SM_XW_BYTES + SM_TOK_BYTES + SM_LEN_BYTES + SM_H_BYTES)

union AF { v16h v; unsigned u[8]; };

__global__ __launch_bounds__(256, 1)
void rnn_scan_kernel(const int* __restrict__ x_in,
                     const int* __restrict__ x_len,
                     const float* __restrict__ xw_g,
                     const _Float16* __restrict__ whh_packed,
                     float* __restrict__ last_h) {
    extern __shared__ char smem[];
    float*    lds_xw  = (float*)smem;
    int*      lds_tok = (int*)(smem + SM_XW_BYTES);
    int*      lds_len = (int*)(smem + SM_XW_BYTES + SM_TOK_BYTES);
    _Float16* lds_h   = (_Float16*)(smem + SM_XW_BYTES + SM_TOK_BYTES + SM_LEN_BYTES);
    unsigned* lds_h_u = (unsigned*)lds_h;

    const int tid  = threadIdx.x;
    const int wave = tid >> 5;
    const int lane = tid & 31;
    const int b0   = blockIdx.x * 16;

    // Stage xw table (f32), token stream [m][t], lengths; zero h buffer 0.
    for (int i = tid; i < V_ * H_; i += 256) lds_xw[i] = xw_g[i];
    for (int i = tid; i < 16 * S_; i += 256) {
        const int m = i / S_, t = i - m * S_;
        lds_tok[i] = x_in[(b0 + m) * S_ + t];
    }
    if (tid < 16) lds_len[tid] = x_len[b0 + tid];
    for (int i = tid; i < (16 * H_) / 2; i += 256) lds_h_u[i] = 0u;

    // Resident B fragments for this wave's 32 output columns: 2 x 8 x v16h.
    const int ntile0 = wave * 2;
    v16h bfrag[2][8];
#pragma unroll
    for (int nt = 0; nt < 2; ++nt)
#pragma unroll
        for (int kt = 0; kt < 8; ++kt)
            bfrag[nt][kt] = *(const v16h*)(whh_packed +
                              (((ntile0 + nt) * 8 + kt) * 512 + lane * 16));

    __syncthreads();

    const int m_lane = lane & 15;                 // A-matrix row for this lane
    const int hi8    = (lane >= 16) ? 8 : 0;      // K / M offset for upper half-wave
    const int n0     = ntile0 * 16 + m_lane;      // first output column

    // Loop-invariant per-lane sequence lengths (C layout rows m = i + hi8).
    int lenreg[8];
#pragma unroll
    for (int i = 0; i < 8; ++i) lenreg[i] = lds_len[i + hi8];

    for (int t = 0; t < S_; ++t) {
        const int cur = t & 1;
        const int nxt = cur ^ 1;
        v8f acc0 = {};
        v8f acc1 = {};

        // Issue the 8 token gathers up-front so they overlap with the GEMM.
        int toks[8];
#pragma unroll
        for (int i = 0; i < 8; ++i) toks[i] = lds_tok[(i + hi8) * S_ + t];

        // u32 view of this lane's h row (2 f16 per u32), current buffer.
        const unsigned* hrow = lds_h_u + ((cur * 16 + m_lane) * H_) / 2;

        // A-fragment layout: vgpr r -> K = 2*(r%4) + 16*(r/4) + 8*(lane>=16) (+kt*32).
        // 2-deep software pipeline over the 8 K-tiles.
        AF af[2];
#pragma unroll
        for (int r = 0; r < 8; ++r) {
            const int kb = ((r & 3) << 1) + ((r >> 2) << 4) + hi8;
            af[0].u[r] = hrow[kb >> 1];
        }
#pragma unroll
        for (int kt = 0; kt < 8; ++kt) {
            if (kt < 7) {
#pragma unroll
                for (int r = 0; r < 8; ++r) {
                    const int kb = ((r & 3) << 1) + ((r >> 2) << 4) + hi8 + ((kt + 1) << 5);
                    af[(kt + 1) & 1].u[r] = hrow[kb >> 1];
                }
            }
            acc0 = __builtin_amdgcn_wmma_f32_16x16x32_f16(
                       false, af[kt & 1].v, false, bfrag[0][kt], (short)0, acc0, false, false);
            acc1 = __builtin_amdgcn_wmma_f32_16x16x32_f16(
                       false, af[kt & 1].v, false, bfrag[1][kt], (short)0, acc1, false, false);
        }

        // Batched xw gather: issue all 16 LDS loads, then consume.
        float xv0[8], xv1[8];
#pragma unroll
        for (int i = 0; i < 8; ++i) {
            const float* xr = lds_xw + toks[i] * H_ + n0;
            xv0[i] = xr[0];
            xv1[i] = xr[16];
        }

        // tanh, f16 writeback, last-hidden capture.
        _Float16* hn = lds_h + nxt * 16 * H_;
#pragma unroll
        for (int i = 0; i < 8; ++i) {
            const int m = i + hi8;
            const float h0 = tanhf(acc0[i] + xv0[i]);
            const float h1 = tanhf(acc1[i] + xv1[i]);
            hn[m * H_ + n0]      = (_Float16)h0;
            hn[m * H_ + n0 + 16] = (_Float16)h1;
            if (lenreg[i] == t + 1) {
                last_h[(b0 + m) * H_ + n0]      = h0;
                last_h[(b0 + m) * H_ + n0 + 16] = h1;
            }
        }
        __syncthreads();
    }
}

// ---------------------------------------------------------------------------
// Kernel 4: classifier head (fp32; 139 MFLOP total -> negligible).
// ---------------------------------------------------------------------------
__global__ void head_kernel(const float* __restrict__ last_h,
                            const float* __restrict__ fc1_w,
                            const float* __restrict__ fc1_b,
                            const float* __restrict__ fc2_w,
                            const float* __restrict__ fc2_b,
                            float* __restrict__ out) {
    __shared__ float sh[H_];
    __shared__ float sy[H_];
    const int b = blockIdx.x;
    const int j = threadIdx.x;
    sh[j] = last_h[b * H_ + j];
    __syncthreads();
    float acc = fc1_b[j];
    const float* w1 = fc1_w + j * H_;
#pragma unroll 4
    for (int k = 0; k < H_; ++k) acc += sh[k] * w1[k];
    sy[j] = fmaxf(acc, 0.0f);
    __syncthreads();
    if (j < C_) {
        float a2 = fc2_b[j];
        const float* w2 = fc2_w + j * H_;
#pragma unroll 4
        for (int k = 0; k < H_; ++k) a2 += sy[k] * w2[k];
        out[b * C_ + j] = a2;
    }
}

// ---------------------------------------------------------------------------
extern "C" void kernel_launch(void* const* d_in, const int* in_sizes, int n_in,
                              void* d_out, int out_size, void* d_ws, size_t ws_size,
                              hipStream_t stream) {
    (void)in_sizes; (void)n_in; (void)out_size; (void)ws_size;
    const int*   x_in  = (const int*)d_in[0];
    const int*   x_len = (const int*)d_in[1];
    const float* emb   = (const float*)d_in[2];
    const float* W_ih  = (const float*)d_in[3];
    const float* W_hh  = (const float*)d_in[4];
    const float* b_ih  = (const float*)d_in[5];
    const float* b_hh  = (const float*)d_in[6];
    const float* fc1_w = (const float*)d_in[7];
    const float* fc1_b = (const float*)d_in[8];
    const float* fc2_w = (const float*)d_in[9];
    const float* fc2_b = (const float*)d_in[10];
    float* out = (float*)d_out;

    // Workspace layout (deterministic, fully overwritten each call):
    char* ws = (char*)d_ws;
    float*    xw     = (float*)ws;                         // 128 KB
    _Float16* whh_pk = (_Float16*)(ws + 131072);           // 128 KB
    float*    lasth  = (float*)(ws + 262144);              //   1 MB

    xw_kernel<<<V_, H_, 0, stream>>>(emb, W_ih, b_ih, b_hh, xw);
    pack_whh_kernel<<<(H_ * H_) / 256, 256, 0, stream>>>(W_hh, whh_pk);
    rnn_scan_kernel<<<B_ / 16, 256, SM_TOTAL, stream>>>(x_in, x_len, xw, whh_pk, lasth);
    head_kernel<<<B_, H_, 0, stream>>>(lasth, fc1_w, fc1_b, fc2_w, fc2_b, out);
}
